// NearestNeighborMatcher_88330297409772
// MI455X (gfx1250) — matrix operand order repaired and checked by reading, not compiled, not run on hardware
//
#include <hip/hip_runtime.h>

typedef _Float16 v16h __attribute__((ext_vector_type(16)));
typedef _Float16 v8h  __attribute__((ext_vector_type(8)));
typedef float    v8f  __attribute__((ext_vector_type(8)));
typedef unsigned int u32x4 __attribute__((ext_vector_type(4)));
typedef int          i32x8 __attribute__((ext_vector_type(8)));
typedef int          i32x4 __attribute__((ext_vector_type(4)));

#define D_DIM      256
#define LDS_STRIDE 264            // 256 halfs + 16B TDM pad -> 528B row stride
#define ROWS_WG     64            // 4 waves x 16 rows
#define TILE_M      32            // columns per TDM tile (2 WMMA subtiles)
#define THREADS    128
#define BUF_HALFS  (TILE_M * LDS_STRIDE)
#define BUF_BYTES  (BUF_HALFS * 2)

// ---------------------------------------------------------------------------
// Prepass: [B, D, N] f32  ->  [B, N, D] f16 (descriptor-major), LDS-tiled
// transpose so both global read and write are coalesced.
// ---------------------------------------------------------------------------
__global__ __launch_bounds__(256)
void convert_transpose_kernel(const float* __restrict__ src,  // [B, D, N]
                              _Float16* __restrict__ dst,     // [B, N, D]
                              int N)
{
    __shared__ float tile[64][65];
    const int n0 = blockIdx.x * 64;
    const int d0 = blockIdx.y * 64;
    const int b  = blockIdx.z;
    const int t  = threadIdx.x;

    #pragma unroll
    for (int i = 0; i < 16; ++i) {
        int idx = t + 256 * i;
        int n  = idx & 63;
        int dd = idx >> 6;
        tile[dd][n] = src[((size_t)(b * D_DIM + d0 + dd)) * (size_t)N + n0 + n];
    }
    __syncthreads();
    #pragma unroll
    for (int i = 0; i < 16; ++i) {
        int idx = t + 256 * i;
        int dd = idx & 63;
        int n  = idx >> 6;
        dst[((size_t)b * N + n0 + n) * (size_t)D_DIM + d0 + dd] = (_Float16)tile[dd][n];
    }
}

// ---------------------------------------------------------------------------
// TDM issue: DMA a TILE_M x 256 f16 tile (rows contiguous, 512B each) into
// LDS at lds_off, padding 16B after every 512B row -> 528B LDS row stride.
// D# built per CDNA5 ISA ch.8 (2-D tensor, groups 2/3 zero).
// ---------------------------------------------------------------------------
__device__ __forceinline__ void tdm_issue_tile(unsigned lds_off, unsigned long long gaddr)
{
    u32x4 g0;
    g0[0] = 1u;                                    // count=1, user descriptor
    g0[1] = lds_off;                               // lds_addr (bytes)
    g0[2] = (unsigned)(gaddr & 0xFFFFFFFFull);     // global_addr[31:0]
    g0[3] = (unsigned)((gaddr >> 32) & 0x1FFFFFFull) | (2u << 30);  // addr[56:32] | type=2

    i32x8 g1;
    g1[0] = (int)((1u << 16) |                     // data_size = 2 bytes
                  (1u << 20) |                     // pad_enable
                  (6u << 22) |                     // pad_interval: every 128 dwords (512B)
                  (3u << 25));                     // pad_amount: 4 dwords (16B)
    g1[1] = (int)(((unsigned)D_DIM) << 16);        // tensor_dim0[15:0] = 256
    g1[2] = (int)(((unsigned)TILE_M) << 16);       // tensor_dim1[15:0] = 32
    g1[3] = (int)(((unsigned)D_DIM) << 16);        // tile_dim0 = 256
    g1[4] = (int)TILE_M;                           // tile_dim1 = 32, tile_dim2 = 0
    g1[5] = D_DIM;                                 // tensor_dim0_stride = 256 (halfs)
    g1[6] = 0;
    g1[7] = 0;

    i32x4 gz4 = {0, 0, 0, 0};
    i32x8 gz8 = {0, 0, 0, 0, 0, 0, 0, 0};
    __builtin_amdgcn_tensor_load_to_lds(g0, g1, gz4, gz4, gz8, 0);
}

// ---------------------------------------------------------------------------
// Fused GEMM + row-wise top-2 + ratio test. A panel lives in registers in
// WMMA layout; B tiles streamed by the Tensor Data Mover, double-buffered.
// ---------------------------------------------------------------------------
__global__ __launch_bounds__(THREADS)
void nn_top2_tdm_kernel(const _Float16* __restrict__ dRowH,  // [B, N, D] f16
                        const _Float16* __restrict__ dColH,  // [B, M, D] f16
                        float* __restrict__ scores_out,      // [B, N]
                        float* __restrict__ matches_f,       // [B, N]
                        int*   __restrict__ matches_i,       // [B, N]
                        int N, int M)
{
    __shared__ __align__(16) _Float16 lB[2 * BUF_HALFS];

    const int b    = blockIdx.y;
    const int n0   = blockIdx.x * ROWS_WG;
    const int t    = threadIdx.x;
    const int lane = t & 31;
    const int w    = t >> 5;
    const int lr   = lane & 15;
    const int h    = lane >> 4;

    // ---- A panel: 16 rows x 256 K per wave, straight into WMMA A-layout regs.
    //  lane holds row M=lr; K-step kb chunks [kb+8h..+7] and [kb+16+8h..+7].
    const _Float16* arow =
        dRowH + ((size_t)b * N + n0 + w * 16 + lr) * (size_t)D_DIM + 8 * h;
    v16h af[8];
    #pragma unroll
    for (int k8 = 0; k8 < 8; ++k8) {
        v8h lo = *(const v8h*)(arow + 32 * k8);
        v8h hi = *(const v8h*)(arow + 32 * k8 + 16);
        af[k8] = __builtin_shufflevector(lo, hi,
            0, 1, 2, 3, 4, 5, 6, 7, 8, 9, 10, 11, 12, 13, 14, 15);
    }

    const unsigned lds_base = (unsigned)(uintptr_t)lB;
    const unsigned long long colbase =
        (unsigned long long)(uintptr_t)(dColH + (size_t)b * M * D_DIM);

    float bst1[8], bst2[8];
    int   id1[8];
    #pragma unroll
    for (int v = 0; v < 8; ++v) { bst1[v] = -1e30f; bst2[v] = -1e30f; id1[v] = 0; }

    const int NT = M / TILE_M;

    // Prologue: DMA tile 0, wait, share.
    if (w == 0) {
        tdm_issue_tile(lds_base, colbase);
        __builtin_amdgcn_s_wait_tensorcnt((short)0);
    }
    __syncthreads();

    for (int i = 0; i < NT; ++i) {
        const _Float16* cur = lB + (i & 1) * BUF_HALFS;
        // Kick off next tile's DMA; overlaps with this tile's WMMAs.
        if (w == 0 && (i + 1) < NT) {
            tdm_issue_tile(lds_base + (unsigned)(((i + 1) & 1) * BUF_BYTES),
                           colbase + (unsigned long long)(i + 1) * TILE_M * (D_DIM * 2));
        }

        v8f acc0 = {};
        v8f acc1 = {};
        const _Float16* pb = cur + lr * LDS_STRIDE + 8 * h;
        #pragma unroll
        for (int k8 = 0; k8 < 8; ++k8) {
            const int kb = 32 * k8;
            v8h b0l = *(const v8h*)(pb + kb);
            v8h b0h = *(const v8h*)(pb + kb + 16);
            v16h bf0 = __builtin_shufflevector(b0l, b0h,
                0, 1, 2, 3, 4, 5, 6, 7, 8, 9, 10, 11, 12, 13, 14, 15);
            acc0 = __builtin_amdgcn_wmma_f32_16x16x32_f16(
                false, af[k8], false, bf0, (short)0, acc0, false, false);

            v8h b1l = *(const v8h*)(pb + 16 * LDS_STRIDE + kb);
            v8h b1h = *(const v8h*)(pb + 16 * LDS_STRIDE + kb + 16);
            v16h bf1 = __builtin_shufflevector(b1l, b1h,
                0, 1, 2, 3, 4, 5, 6, 7, 8, 9, 10, 11, 12, 13, 14, 15);
            acc1 = __builtin_amdgcn_wmma_f32_16x16x32_f16(
                false, af[k8], false, bf1, (short)0, acc1, false, false);
        }

        // Fold the two 16x16 result tiles into the running top-2.
        const int m0   = i * TILE_M;
        const int col0 = m0 + lr;
        const int col1 = m0 + 16 + lr;
        #pragma unroll
        for (int v = 0; v < 8; ++v) {
            float s0 = acc0[v];
            if (s0 > bst1[v]) { bst2[v] = bst1[v]; bst1[v] = s0; id1[v] = col0; }
            else if (s0 > bst2[v]) { bst2[v] = s0; }
            float s1 = acc1[v];
            if (s1 > bst1[v]) { bst2[v] = bst1[v]; bst1[v] = s1; id1[v] = col1; }
            else if (s1 > bst2[v]) { bst2[v] = s1; }
        }

        __syncthreads();   // everyone done reading `cur` (and prior `next`)
        if (w == 0 && (i + 1) < NT)
            __builtin_amdgcn_s_wait_tensorcnt((short)0);   // next tile landed
        __syncthreads();
    }

    // ---- Merge top-2 across the 16 lanes sharing each row, ratio test, write.
    #pragma unroll
    for (int v = 0; v < 8; ++v) {
        float x1 = bst1[v], x2 = bst2[v];
        int   xi = id1[v];
        #pragma unroll
        for (int off = 1; off < 16; off <<= 1) {
            float o1 = __shfl_xor(x1, off, 32);
            float o2 = __shfl_xor(x2, off, 32);
            int   oi = __shfl_xor(xi, off, 32);
            bool take = o1 > x1;
            float n2  = take ? fmaxf(x1, o2) : fmaxf(o1, x2);
            xi = take ? oi : xi;
            x1 = take ? o1 : x1;
            x2 = n2;
        }
        if (lr == 0) {
            int nrow = n0 + w * 16 + v + 8 * h;
            float dd1 = 2.0f * (1.0f - x1);
            float dd2 = 2.0f * (1.0f - x2);
            bool ok = dd1 <= 0.64f * dd2;          // RATIO_THRESH^2
            int  mi = ok ? xi : -1;
            size_t g = (size_t)b * (size_t)N + nrow;
            matches_i[g]  = mi;
            matches_f[g]  = (float)mi;
            scores_out[g] = ok ? (x1 + 1.0f) * 0.5f : 0.0f;
        }
    }
}

__global__ void mutual_check_kernel(const int* __restrict__ m0i,
                                    const int* __restrict__ m1i,
                                    float* __restrict__ m0f,
                                    int N, int M, int total)
{
    int g = blockIdx.x * blockDim.x + threadIdx.x;
    if (g >= total) return;
    int b = g / N;
    int n = g - b * N;
    int m = m0i[g];
    bool ok = (m > -1) && (m1i[(size_t)b * (size_t)M + m] == n);
    m0f[g] = ok ? (float)m : -1.0f;
}

extern "C" void kernel_launch(void* const* d_in, const int* in_sizes, int n_in,
                              void* d_out, int out_size, void* d_ws, size_t ws_size,
                              hipStream_t stream) {
    (void)in_sizes; (void)n_in; (void)out_size; (void)ws_size;
    const float* d0 = (const float*)d_in[0];   // [B, D, N]
    const float* d1 = (const float*)d_in[1];   // [B, D, M]
    const int B = 4, N = 4096, M = 4096;

    // Output tuple: matches0[B*N], matches1[B*M], scores0[B*N], scores1[B*M]
    float* out = (float*)d_out;
    float* m0f = out;
    float* m1f = out + (size_t)B * N;
    float* s0  = m1f + (size_t)B * M;
    float* s1  = s0  + (size_t)B * N;

    // Workspace: f16 transposed descriptor copies + int match scratch.
    _Float16* d0h = (_Float16*)d_ws;                       // [B, N, D]
    _Float16* d1h = d0h + (size_t)B * N * D_DIM;           // [B, M, D]
    int* m0i = (int*)(d1h + (size_t)B * M * D_DIM);
    int* m1i = m0i + B * N;

    convert_transpose_kernel<<<dim3(N / 64, D_DIM / 64, B), 256, 0, stream>>>(d0, d0h, N);
    convert_transpose_kernel<<<dim3(M / 64, D_DIM / 64, B), 256, 0, stream>>>(d1, d1h, M);

    dim3 blk(THREADS);
    nn_top2_tdm_kernel<<<dim3(N / ROWS_WG, B), blk, 0, stream>>>(d0h, d1h, s0, m0f, m0i, N, M);
    nn_top2_tdm_kernel<<<dim3(M / ROWS_WG, B), blk, 0, stream>>>(d1h, d0h, s1, m1f, m1i, M, N);

    int total = B * N;
    mutual_check_kernel<<<(total + 255) / 256, 256, 0, stream>>>(m0i, m1i, m0f, N, M, total);
}
